// GroupQueryAttention_38319698215587
// MI455X (gfx1250) — matrix operand order repaired and checked by reading, not compiled, not run on hardware
//
#include <hip/hip_runtime.h>
#include <hip/hip_bf16.h>
#include <stdint.h>

// ---------------------------------------------------------------------------
// GQA forward for MI455X (gfx1250): bf16 WMMA + async global->LDS (ASYNCcnt),
// double-buffered GEMM tiles, flash attention.
// ---------------------------------------------------------------------------

typedef __attribute__((ext_vector_type(16))) __bf16 bf16x16;
typedef __attribute__((ext_vector_type(8)))  __bf16 bf16x8;
typedef __attribute__((ext_vector_type(8)))  float  f32x8;
typedef __attribute__((vector_size(16)))     int    v4i_;

#define D_MODEL   2048
#define QKV_N     3072
#define KV_DIM    512
#define N_HEAD    32
#define N_GROUPS  8
#define HEAD_DIM  64
#define SEQ_T     2048
#define BATCH     2
#define M_TOT     (BATCH * SEQ_T)   // 4096

#define AS_STR    40                // LDS row stride (bf16) for 32-wide K tiles
#define GBUF      (128 * AS_STR)    // one GEMM LDS buffer (elems)
#define KT_STR    72                // attention K-tile LDS row stride
#define VT_STR    40                // transposed V-tile LDS row stride

#if __has_builtin(__builtin_amdgcn_global_load_async_to_lds_b128)
#define HAVE_ASYNC_LDS 1
#else
#define HAVE_ASYNC_LDS 0
#endif

__device__ __forceinline__ __bf16 f2bf(float f) {
  uint32_t u = __builtin_bit_cast(uint32_t, f);
  u += 0x7FFFu + ((u >> 16) & 1u);            // round-to-nearest-even
  uint16_t h = (uint16_t)(u >> 16);
  return __builtin_bit_cast(__bf16, h);
}

// s_wait_asynccnt with immediate; memory clobber pins LDS reads behind it.
template <int N>
__device__ __forceinline__ void wait_asynccnt() {
  asm volatile("s_wait_asynccnt %0" :: "n"(N) : "memory");
}

// One 16-byte global->LDS copy. Async DMA path on gfx1250 (no VGPR staging),
// plain load/store fallback otherwise.
__device__ __forceinline__ void async_b128(const __bf16* g, __bf16* s) {
#if HAVE_ASYNC_LDS
  __builtin_amdgcn_global_load_async_to_lds_b128(
      (__attribute__((address_space(1))) v4i_*)(uintptr_t)g,
      (__attribute__((address_space(3))) v4i_*)(uint32_t)(uintptr_t)s,
      0, 0);
#else
  *(float4*)s = *(const float4*)g;
#endif
}

__device__ __forceinline__ bf16x16 frag_combine(const __bf16* p0, const __bf16* p1) {
  bf16x8 a = *(const bf16x8*)p0;
  bf16x8 b = *(const bf16x8*)p1;
  bf16x16 r;
#pragma unroll
  for (int i = 0; i < 8; ++i) { r[i] = a[i]; r[i + 8] = b[i]; }
  return r;
}

// A-frag (16x32 bf16, row-major src): lane<16 -> row=lane, K {0..7,16..23};
// lane>=16 -> row=lane-16, K {8..15,24..31}.
__device__ __forceinline__ bf16x16 loadA(const __bf16* base, int stride, int lane) {
  const int half = lane >> 4, l = lane & 15;
  const __bf16* p = base + l * stride + half * 8;
  return frag_combine(p, p + 16);
}

// B-frag (32x16 bf16) from K-major storage (row = N col, elems along K):
// lane<16 -> col=lane, K {0..7,16..23}; lane>=16 -> col=lane-16, K {8..15,24..31}.
__device__ __forceinline__ bf16x16 loadB(const __bf16* base, int stride, int lane) {
  const int half = lane >> 4, l = lane & 15;
  const __bf16* p = base + l * stride + half * 8;
  return frag_combine(p, p + 16);
}

// ---------------------------------------------------------------------------
// 128x128 GEMM core, double-buffered 32-wide K tiles filled by async DMA.
// A [M][K] bf16 row-major, Bt [N][K] bf16 (K-major).
// 8 waves in 4(M) x 2(N); each wave owns a 32x64 tile (2x4 WMMA accumulators).
// Per thread per tile: 4 async b128 copies -> wait threshold 4.
// ---------------------------------------------------------------------------
__device__ __forceinline__ void gemm_core_128x128(
    const __bf16* __restrict__ A, const __bf16* __restrict__ Bt,
    int m0, int n0, f32x8 acc[2][4], __bf16* As, __bf16* Bs)
{
  const int tid = threadIdx.x, lane = tid & 31, wave = tid >> 5;
  const int wm = wave & 3, wn = wave >> 2;
  const int ldr = tid >> 1, ldk = (tid & 1) * 16;

  const __bf16* gA = A + (size_t)(m0 + ldr) * D_MODEL + ldk;
  const __bf16* gB = Bt + (size_t)(n0 + ldr) * D_MODEL + ldk;
  __bf16* sA = As + ldr * AS_STR + ldk;
  __bf16* sB = Bs + ldr * AS_STR + ldk;

  // prologue: tile 0 -> buffer 0
  async_b128(gA, sA);
  async_b128(gA + 8, sA + 8);
  async_b128(gB, sB);
  async_b128(gB + 8, sB + 8);

#pragma unroll 1
  for (int k0 = 0; k0 < D_MODEL; k0 += 32) {
    const int cur = (k0 >> 5) & 1;
    const int nxt = cur ^ 1;
    if (k0 + 32 < D_MODEL) {            // fill next tile into other buffer
      async_b128(gA + k0 + 32, sA + nxt * GBUF);
      async_b128(gA + k0 + 40, sA + nxt * GBUF + 8);
      async_b128(gB + k0 + 32, sB + nxt * GBUF);
      async_b128(gB + k0 + 40, sB + nxt * GBUF + 8);
      wait_asynccnt<4>();               // in-order: current tile landed
    } else {
      wait_asynccnt<0>();
    }
    __syncthreads();                    // all threads' tile data visible

    const __bf16* cA = As + cur * GBUF;
    const __bf16* cB = Bs + cur * GBUF;
    bf16x16 af[2];
#pragma unroll
    for (int mt = 0; mt < 2; ++mt)
      af[mt] = loadA(cA + (wm * 32 + mt * 16) * AS_STR, AS_STR, lane);
#pragma unroll
    for (int nt = 0; nt < 4; ++nt) {
      bf16x16 bfv = loadB(cB + (wn * 64 + nt * 16) * AS_STR, AS_STR, lane);
#pragma unroll
      for (int mt = 0; mt < 2; ++mt)
        acc[mt][nt] = __builtin_amdgcn_wmma_f32_16x16x32_bf16(
            false, af[mt], false, bfv, (short)0, acc[mt][nt], false, false);
    }
    __syncthreads();                    // consume-done before buffer reuse
  }
}

// ---------------------------------------------------------------------------
// QKV projection + fused RoPE, scatters to Q[B,H,T,64], K/V[B,G,T,64] bf16.
// ---------------------------------------------------------------------------
__global__ __launch_bounds__(256)
void gqa_qkv_rope_kernel(const __bf16* __restrict__ xb, const __bf16* __restrict__ wqkvT,
                         __bf16* __restrict__ qb, __bf16* __restrict__ kbuf,
                         __bf16* __restrict__ vbuf)
{
  __shared__ __bf16 As[2 * GBUF];
  __shared__ __bf16 Bs[2 * GBUF];

  const int tid = threadIdx.x, lane = tid & 31, wave = tid >> 5;
  const int wm = wave & 3, wn = wave >> 2;
  const int m0 = blockIdx.y * 128, n0 = blockIdx.x * 128;

  f32x8 acc[2][4];
#pragma unroll
  for (int i = 0; i < 2; ++i)
#pragma unroll
    for (int j = 0; j < 4; ++j)
#pragma unroll
      for (int e = 0; e < 8; ++e) acc[i][j][e] = 0.0f;

  gemm_core_128x128(xb, wqkvT, m0, n0, acc, As, Bs);

  const float inv64 = 1.0f / 64.0f;
#pragma unroll
  for (int mt = 0; mt < 2; ++mt)
#pragma unroll
  for (int nt = 0; nt < 4; ++nt) {
    const int ncol = n0 + wn * 64 + nt * 16 + (lane & 15);
    const int d = ncol & 63;
    // theta depends only on this lane's column pair
    const float theta = __powf(10000.0f, -(float)(d & ~1) * inv64);
#pragma unroll
    for (int r = 0; r < 8; ++r) {
      const int mrow = m0 + wm * 32 + mt * 16 + ((lane >> 4) << 3) + r;
      const int b = mrow >> 11;
      const int t = mrow & (SEQ_T - 1);
      const float v = acc[mt][nt][r];
      // even/odd rotary pairs live in adjacent lanes of the C/D layout
      const float partner = __shfl_xor(v, 1, 32);
      if (ncol < D_MODEL) {
        const int h = ncol >> 6;
        float s, c;
        __sincosf((float)t * theta, &s, &c);
        float out = (d & 1) ? fmaf(partner, s, v * c) : fmaf(-partner, s, v * c);
        qb[(((size_t)b * N_HEAD + h) * SEQ_T + t) * HEAD_DIM + d] = f2bf(out);
      } else if (ncol < D_MODEL + KV_DIM) {
        const int g = (ncol - D_MODEL) >> 6;
        float s, c;
        __sincosf((float)t * theta, &s, &c);
        float out = (d & 1) ? fmaf(partner, s, v * c) : fmaf(-partner, s, v * c);
        kbuf[(((size_t)b * N_GROUPS + g) * SEQ_T + t) * HEAD_DIM + d] = f2bf(out);
      } else {
        const int g = (ncol - D_MODEL - KV_DIM) >> 6;
        vbuf[(((size_t)b * N_GROUPS + g) * SEQ_T + t) * HEAD_DIM + d] = f2bf(v);
      }
    }
  }
}

// ---------------------------------------------------------------------------
// Flash attention: 1 wave = 16 query rows, wave-private LDS, 32-key blocks,
// online softmax, causal mask. grid = (T/64, B*N_HEAD), 128 threads.
// ---------------------------------------------------------------------------
__global__ __launch_bounds__(128)
void gqa_flash_attn_kernel(const __bf16* __restrict__ qb, const __bf16* __restrict__ kbuf,
                           const __bf16* __restrict__ vbuf, __bf16* __restrict__ yb)
{
  __shared__ __bf16 Kt[4][32 * KT_STR];   // K rows, row-major == B-frag layout
  __shared__ __bf16 Vt[4][64 * VT_STR];   // V transposed [d][key]
  __shared__ __bf16 Pt[4][16 * 32];       // softmax probs, A-frag layout

  const int tid = threadIdx.x, lane = tid & 31, wave = tid >> 5;
  const int bh = blockIdx.y;              // b*32 + h
  const int b = bh >> 5, h = bh & 31;
  const int g = h >> 2;                   // 4 heads per KV group
  const int qrow = blockIdx.x * 64 + wave * 16;

  const __bf16* Qp = qb + ((size_t)bh * SEQ_T + qrow) * HEAD_DIM;
  const __bf16* Kp = kbuf + ((size_t)(b * N_GROUPS + g) * SEQ_T) * HEAD_DIM;
  const __bf16* Vp = vbuf + ((size_t)(b * N_GROUPS + g) * SEQ_T) * HEAD_DIM;

  // Q fragments held in registers for the whole key loop (2 k-steps of 32)
  bf16x16 qf[2];
  {
    const int half = lane >> 4, l = lane & 15;
    const __bf16* p = Qp + l * HEAD_DIM + half * 8;
    qf[0] = frag_combine(p, p + 16);
    qf[1] = frag_combine(p + 32, p + 48);
  }

  float m_run[8], l_run[8];
  f32x8 o[4];
#pragma unroll
  for (int r = 0; r < 8; ++r) { m_run[r] = -3.0e38f; l_run[r] = 0.0f; }
#pragma unroll
  for (int dt = 0; dt < 4; ++dt)
#pragma unroll
    for (int r = 0; r < 8; ++r) o[dt][r] = 0.0f;

  __bf16* kt = Kt[wave];
  __bf16* vt = Vt[wave];
  __bf16* pt = Pt[wave];

  const float scale = 0.125f;             // HEAD_DIM^-0.5
  const int kend = qrow + 16;             // exclusive causal key limit
  const int kcol = lane & 15;
  const int half = lane >> 4;

#pragma unroll 1
  for (int kb0 = 0; kb0 < kend; kb0 += 32) {
    // K tile: async DMA, one row per lane (wave-private region, no barrier)
    {
      const __bf16* kr = Kp + (size_t)(kb0 + lane) * HEAD_DIM;
#pragma unroll
      for (int c = 0; c < 8; ++c)
        async_b128(kr + c * 8, &kt[lane * KT_STR + c * 8]);
      // V tile: transposed scatter through VGPRs
      const __bf16* vr = Vp + (size_t)(kb0 + lane) * HEAD_DIM;
#pragma unroll
      for (int c = 0; c < 8; ++c) {
        bf16x8 chunk = *(const bf16x8*)(vr + c * 8);
#pragma unroll
        for (int e = 0; e < 8; ++e)
          vt[(c * 8 + e) * VT_STR + lane] = chunk[e];
      }
    }
    wait_asynccnt<0>();                   // this wave's K tile landed

    // S = Q K^T : 2 key-halves x 2 head-dim k-steps
    f32x8 s[2];
#pragma unroll
    for (int nt = 0; nt < 2; ++nt) {
      f32x8 c;
#pragma unroll
      for (int e = 0; e < 8; ++e) c[e] = 0.0f;
      bf16x16 b0 = loadB(&kt[(nt * 16) * KT_STR + 0], KT_STR, lane);
      c = __builtin_amdgcn_wmma_f32_16x16x32_bf16(false, qf[0], false, b0, (short)0, c, false, false);
      bf16x16 b1 = loadB(&kt[(nt * 16) * KT_STR + 32], KT_STR, lane);
      c = __builtin_amdgcn_wmma_f32_16x16x32_bf16(false, qf[1], false, b1, (short)0, c, false, false);
      s[nt] = c;
    }

    // online softmax (row stats reduced over 16-lane halves)
#pragma unroll
    for (int r = 0; r < 8; ++r) {
      const int qr = qrow + half * 8 + r;
      float x0 = s[0][r] * scale;
      float x1 = s[1][r] * scale;
      if (kb0 + kcol > qr)      x0 = -3.0e38f;
      if (kb0 + 16 + kcol > qr) x1 = -3.0e38f;
      float mx = fmaxf(x0, x1);
#pragma unroll
      for (int off = 8; off >= 1; off >>= 1)
        mx = fmaxf(mx, __shfl_xor(mx, off, 32));
      const float mnew = fmaxf(m_run[r], mx);
      const float alpha = __expf(m_run[r] - mnew);
      const float p0 = __expf(x0 - mnew);
      const float p1 = __expf(x1 - mnew);
      float sum = p0 + p1;
#pragma unroll
      for (int off = 8; off >= 1; off >>= 1)
        sum += __shfl_xor(sum, off, 32);
      l_run[r] = l_run[r] * alpha + sum;
      m_run[r] = mnew;
      const int prow = half * 8 + r;
      pt[prow * 32 + kcol]      = f2bf(p0);
      pt[prow * 32 + 16 + kcol] = f2bf(p1);
#pragma unroll
      for (int dt = 0; dt < 4; ++dt) o[dt][r] *= alpha;
    }

    // O += P V
    bf16x16 pf = loadA(pt, 32, lane);
#pragma unroll
    for (int dt = 0; dt < 4; ++dt) {
      bf16x16 vf = loadB(&vt[(dt * 16) * VT_STR], VT_STR, lane);
      o[dt] = __builtin_amdgcn_wmma_f32_16x16x32_bf16(
          false, pf, false, vf, (short)0, o[dt], false, false);
    }
  }

  // epilogue: y[b*T+t][h*64+d] bf16
#pragma unroll
  for (int r = 0; r < 8; ++r) {
    const float inv = 1.0f / l_run[r];
    const int t = qrow + half * 8 + r;
    const size_t row = (size_t)b * SEQ_T + t;
#pragma unroll
    for (int dt = 0; dt < 4; ++dt)
      yb[row * D_MODEL + h * 64 + dt * 16 + kcol] = f2bf(o[dt][r] * inv);
  }
}

// ---------------------------------------------------------------------------
// Output projection: yb @ w_projT -> fp32 out.
// ---------------------------------------------------------------------------
__global__ __launch_bounds__(256)
void gqa_proj_kernel(const __bf16* __restrict__ yb, const __bf16* __restrict__ wprojT,
                     float* __restrict__ out)
{
  __shared__ __bf16 As[2 * GBUF];
  __shared__ __bf16 Bs[2 * GBUF];

  const int tid = threadIdx.x, lane = tid & 31, wave = tid >> 5;
  const int wm = wave & 3, wn = wave >> 2;
  const int m0 = blockIdx.y * 128, n0 = blockIdx.x * 128;

  f32x8 acc[2][4];
#pragma unroll
  for (int i = 0; i < 2; ++i)
#pragma unroll
    for (int j = 0; j < 4; ++j)
#pragma unroll
      for (int e = 0; e < 8; ++e) acc[i][j][e] = 0.0f;

  gemm_core_128x128(yb, wprojT, m0, n0, acc, As, Bs);

#pragma unroll
  for (int mt = 0; mt < 2; ++mt)
#pragma unroll
  for (int nt = 0; nt < 4; ++nt) {
    const int ncol = n0 + wn * 64 + nt * 16 + (lane & 15);
#pragma unroll
    for (int r = 0; r < 8; ++r) {
      const int mrow = m0 + wm * 32 + mt * 16 + ((lane >> 4) << 3) + r;
      out[(size_t)mrow * D_MODEL + ncol] = acc[mt][nt][r];
    }
  }
}

// ---------------------------------------------------------------------------
// Precision-conversion helpers (fp32 -> bf16, optionally transposed K-major).
// ---------------------------------------------------------------------------
__global__ void gqa_cvt_kernel(const float* __restrict__ in, __bf16* __restrict__ out, int n) {
  int i = blockIdx.x * blockDim.x + threadIdx.x;
  const int stride = gridDim.x * blockDim.x;
  for (; i < n; i += stride) out[i] = f2bf(in[i]);
}

__global__ void gqa_cvt_transpose_kernel(const float* __restrict__ in, __bf16* __restrict__ out,
                                         int K, int N) {  // in[K][N] -> out[N][K]
  int i = blockIdx.x * blockDim.x + threadIdx.x;
  const int stride = gridDim.x * blockDim.x;
  const int total = K * N;
  for (; i < total; i += stride) {
    const int k = i / N, n = i - k * N;
    out[(size_t)n * K + k] = f2bf(in[i]);
  }
}

// ---------------------------------------------------------------------------
extern "C" void kernel_launch(void* const* d_in, const int* in_sizes, int n_in,
                              void* d_out, int out_size, void* d_ws, size_t ws_size,
                              hipStream_t stream) {
  (void)in_sizes; (void)n_in; (void)out_size; (void)ws_size;
  const float* x      = (const float*)d_in[0];
  const float* w_qkv  = (const float*)d_in[1];
  const float* w_proj = (const float*)d_in[2];
  float* out = (float*)d_out;

  char* w = (char*)d_ws;
  __bf16* xb     = (__bf16*)w; w += (size_t)M_TOT * D_MODEL * 2;
  __bf16* wqkvT  = (__bf16*)w; w += (size_t)QKV_N * D_MODEL * 2;
  __bf16* wprojT = (__bf16*)w; w += (size_t)D_MODEL * D_MODEL * 2;
  __bf16* qb     = (__bf16*)w; w += (size_t)BATCH * N_HEAD * SEQ_T * HEAD_DIM * 2;
  __bf16* kbuf   = (__bf16*)w; w += (size_t)BATCH * N_GROUPS * SEQ_T * HEAD_DIM * 2;
  __bf16* vbuf   = (__bf16*)w; w += (size_t)BATCH * N_GROUPS * SEQ_T * HEAD_DIM * 2;
  __bf16* yb     = (__bf16*)w;

  gqa_cvt_kernel<<<2048, 256, 0, stream>>>(x, xb, M_TOT * D_MODEL);
  gqa_cvt_transpose_kernel<<<2048, 256, 0, stream>>>(w_qkv, wqkvT, D_MODEL, QKV_N);
  gqa_cvt_transpose_kernel<<<2048, 256, 0, stream>>>(w_proj, wprojT, D_MODEL, D_MODEL);

  gqa_qkv_rope_kernel<<<dim3(QKV_N / 128, M_TOT / 128), 256, 0, stream>>>(
      xb, wqkvT, qb, kbuf, vbuf);

  gqa_flash_attn_kernel<<<dim3(SEQ_T / 64, BATCH * N_HEAD), 128, 0, stream>>>(
      qb, kbuf, vbuf, yb);

  gqa_proj_kernel<<<dim3(D_MODEL / 128, M_TOT / 128), 256, 0, stream>>>(
      yb, wprojT, out);
}
